// Net_45320494907925
// MI455X (gfx1250) — matrix operand order repaired and checked by reading, not compiled, not run on hardware
//
#include <hip/hip_runtime.h>
#include <hip/hip_bf16.h>

// ---------------------------------------------------------------------------
// Net: masked BN -> subm conv1(1->32)+ReLU -> subm conv2(32->64, WMMA f16,
//      fused 2x2 maxpool + NCHW flatten) -> FC1(12544->128, WMMA f16)+ReLU
//      -> FC2(128->10) + log_softmax
// ---------------------------------------------------------------------------

typedef __attribute__((ext_vector_type(16))) _Float16 v16h;
typedef __attribute__((ext_vector_type(8)))  float    v8f;

union Frag16 { v16h h; uint32_t u[8]; uint4 q[2]; };
union H2U    { _Float16 h[2]; uint32_t u; };

static constexpr int    NB   = 4096;
static constexpr int    HW   = 28 * 28;           // 784
static constexpr int    C1   = 32;
static constexpr int    C2   = 64;
static constexpr int    KFC  = 14 * 14 * 64;      // 12544
static constexpr int    NFC  = 128;
static constexpr int    NX   = NB * HW;           // 3,211,264

// workspace layout (bytes)
static constexpr size_t OFF_STATS = 0;                                   // 16 f32
static constexpr size_t OFF_H1    = 256;
static constexpr size_t H1_BYTES  = (size_t)NB * HW * C1 * 2;            // 205.5 MB
static constexpr size_t OFF_HP    = OFF_H1 + H1_BYTES;
static constexpr size_t HP_BYTES  = (size_t)NB * KFC * 2;                // 102.8 MB
static constexpr size_t OFF_FC1   = OFF_HP + HP_BYTES;
static constexpr size_t FC1_BYTES = (size_t)NB * NFC * 2;                // 1 MB
static constexpr size_t OFF_FW1   = OFF_FC1 + FC1_BYTES;
static constexpr size_t FW1_BYTES = (size_t)KFC * NFC * 2;               // 3.2 MB
static constexpr size_t OFF_W2P   = OFF_FW1 + FW1_BYTES;                 // 9216 dwords

// ---------------------------------------------------------------------------
__global__ void k_init_stats(float* stats) {
    if (threadIdx.x < 16) stats[threadIdx.x] = 0.0f;
}

__global__ void k_reduce(const float* __restrict__ x, float* __restrict__ stats) {
    float s = 0.f, ss = 0.f, c = 0.f;
    for (int i = blockIdx.x * blockDim.x + threadIdx.x; i < NX;
         i += gridDim.x * blockDim.x) {
        float v = x[i];
        if (v != 0.f) { s += v; ss += v * v; c += 1.f; }
    }
    for (int off = 16; off > 0; off >>= 1) {
        s  += __shfl_down(s,  off);
        ss += __shfl_down(ss, off);
        c  += __shfl_down(c,  off);
    }
    if ((threadIdx.x & 31) == 0) {
        atomicAdd(&stats[0], s);
        atomicAdd(&stats[1], ss);
        atomicAdd(&stats[2], c);
    }
}

__global__ void k_finalize(float* stats, const float* g, const float* b) {
    float n    = fmaxf(stats[2], 1.0f);
    float mean = stats[0] / n;
    float var  = stats[1] / n - mean * mean;
    float inv  = rsqrtf(var + 1e-5f);
    stats[3] = mean;
    stats[4] = g[0] * inv;   // scale
    stats[5] = b[0];         // beta
}

// fcw1 f32 -> f16 copy (layout unchanged: [128][12544])
__global__ void k_prep_fcw1(const float* __restrict__ w, _Float16* __restrict__ wh) {
    int i = blockIdx.x * blockDim.x + threadIdx.x;
    if (i < KFC * NFC) wh[i] = (_Float16)w[i];
}

// pack w2 [3,3,32,64] HWIO into WMMA B fragments:
// index ((tap*4+nt)*32 + lane)*8 + v holds f16 pair (k,k+1) of column n,
// with k = 2v + 16*(lane>=16), n = nt*16 + (lane&15).
__global__ void k_prep_w2(const float* __restrict__ w2, uint32_t* __restrict__ w2p) {
    int tid = blockIdx.x * blockDim.x + threadIdx.x;
    if (tid >= 9 * 4 * 32 * 8) return;
    int v    = tid & 7;
    int lane = (tid >> 3) & 31;
    int nt   = (tid >> 8) & 3;
    int tap  = tid >> 10;
    int k    = 2 * v + ((lane & 16) ? 16 : 0);
    int n    = nt * 16 + (lane & 15);
    H2U p;
    p.h[0] = (_Float16)w2[(tap * 32 + k    ) * 64 + n];
    p.h[1] = (_Float16)w2[(tap * 32 + k + 1) * 64 + n];
    w2p[tid] = p.u;
}

// masked BN + conv1 (1->32) + bias + mask + ReLU, store h1 f16 [b][site][c]
__global__ __launch_bounds__(128)
void k_bn_conv1(const float* __restrict__ x, const float* __restrict__ stats,
                const float* __restrict__ w1, const float* __restrict__ b1,
                _Float16* __restrict__ h1) {
    int idx = blockIdx.x * blockDim.x + threadIdx.x;
    if (idx >= NX) return;
    int b = idx / HW, s = idx % HW;
    int y = s / 28, xx = s % 28;
    float mean = stats[3], scale = stats[4], betav = stats[5];
    float cx   = x[idx];
    bool  act  = (cx != 0.f);
    float ft[9];
    #pragma unroll
    for (int t = 0; t < 9; ++t) {
        int dy = t / 3 - 1, dx = t % 3 - 1;
        int ny = y + dy, nx = xx + dx;
        float v = 0.f;
        if (ny >= 0 && ny < 28 && nx >= 0 && nx < 28) {
            float xv = x[(size_t)b * HW + ny * 28 + nx];
            if (xv != 0.f) v = scale * (xv - mean) + betav;
        }
        ft[t] = v;
    }
    _Float16* out = h1 + (size_t)idx * C1;
    for (int c = 0; c < C1; ++c) {
        float a = b1[c];
        #pragma unroll
        for (int t = 0; t < 9; ++t) a += ft[t] * w1[t * C1 + c];
        out[c] = act ? (_Float16)fmaxf(a, 0.f) : (_Float16)0.f;
    }
}

// conv2 (32->64) as implicit GEMM via v_wmma_f32_16x16x32_f16, with the 2x2
// maxpool + NCHW flatten fused into the epilogue (h2 never hits HBM).
// One block per image. M-tile = 2 rows x 8 cols so every pool window is
// wave-local: vertical max via __shfl_xor(,16), horizontal max in-lane.
// Image staged into padded 30x30x32 f16 LDS via async global->LDS b128.
__global__ __launch_bounds__(128)
void k_conv2(const float* __restrict__ x, const _Float16* __restrict__ h1,
             const uint32_t* __restrict__ w2p, const float* __restrict__ b2,
             _Float16* __restrict__ hp) {
    __shared__ uint32_t      simg[30 * 30 * 16];   // 57.6 KB (f16 pairs)
    __shared__ unsigned char smask[816];
    int b = blockIdx.x, tid = threadIdx.x;
    int lane = tid & 31, wv = tid >> 5;

    for (int t = tid; t < 30 * 30 * 16; t += 128) simg[t] = 0;
    for (int t = tid; t < 816; t += 128) smask[t] = 0;
    __syncthreads();

    // async DMA of the image interior: 3136 x 16B chunks, global -> LDS
    {
        const char* gsrc = (const char*)(h1 + (size_t)b * HW * C1);
        uint32_t lds_base = (uint32_t)(uintptr_t)(&simg[0]);
        for (int t = tid; t < 3136; t += 128) {
            int site = t >> 2, part = t & 3;
            int y = site / 28, xx = site % 28;
            uint32_t ld = lds_base +
                (uint32_t)((((y + 1) * 30 + xx + 1) * 64) + part * 16);
            uint64_t ga = (uint64_t)(uintptr_t)(gsrc + (size_t)t * 16);
            asm volatile("global_load_async_to_lds_b128 %0, %1, off"
                         :: "v"(ld), "v"(ga) : "memory");
        }
    }
    for (int t = tid; t < HW; t += 128)
        smask[t] = (x[(size_t)b * HW + t] != 0.f) ? 1 : 0;
    asm volatile("s_wait_asynccnt 0x0" ::: "memory");
    __syncthreads();

    int m  = lane & 15;
    int hb = (lane >> 4) & 1;
    int c0 = hb * 8;                       // A fragment lane-half channel base
    // 56 M-tiles: mt = ry*4 + cg, rows {2ry, 2ry+1}, cols cg*8 .. cg*8+7
    for (int mt = wv; mt < 56; mt += 4) {
        int ry = mt >> 2, cg = mt & 3;
        int y  = ry * 2 + ((m >> 3) & 1);  // A row m -> site row
        int xx = cg * 8 + (m & 7);         // may be >= 28 (discarded rows)
        v8f acc[4] = {v8f{}, v8f{}, v8f{}, v8f{}};
        #pragma unroll
        for (int tap = 0; tap < 9; ++tap) {
            int dy = tap / 3, dx = tap % 3;
            int base = ((y + dy) * 30 + (xx + dx)) * 16 + c0;
            Frag16 a;
            a.q[0] = *(const uint4*)&simg[base];
            a.q[1] = *(const uint4*)&simg[base + 4];
            const uint32_t* bp = w2p + (size_t)tap * 1024 + lane * 8;
            #pragma unroll
            for (int nt = 0; nt < 4; ++nt) {
                Frag16 bf;
                bf.q[0] = *(const uint4*)(bp + nt * 256);
                bf.q[1] = *(const uint4*)(bp + nt * 256 + 4);
                acc[nt] = __builtin_amdgcn_wmma_f32_16x16x32_f16(
                    false, a.h, false, bf.h, (short)0, acc[nt], false, false);
            }
        }
        // epilogue: bias + mask + ReLU + 2x2 maxpool + NCHW-flatten store
        #pragma unroll
        for (int nt = 0; nt < 4; ++nt) {
            int n = nt * 16 + m;
            float bias = b2[n];
            float vm[8];
            #pragma unroll
            for (int r = 0; r < 8; ++r) {
                int yy = ry * 2 + hb;      // this lane's D rows: r + hb*8
                int xs = cg * 8 + r;
                float v = 0.f;
                if (xs < 28 && smask[yy * 28 + xs])
                    v = fmaxf(acc[nt][r] + bias, 0.f);
                vm[r] = fmaxf(v, __shfl_xor(v, 16));   // vertical pool
            }
            if (hb == 0) {
                #pragma unroll
                for (int r = 0; r < 8; r += 2) {
                    int px = cg * 4 + (r >> 1);
                    if (px < 14) {
                        float p = fmaxf(vm[r], vm[r + 1]);  // horizontal pool
                        hp[(size_t)b * KFC + n * 196 + ry * 14 + px] =
                            (_Float16)p;
                    }
                }
            }
        }
    }
}

// FC1: [4096,12544] x [12544,128]^T via WMMA; one wave per 16x16 tile.
__global__ __launch_bounds__(256)
void k_fc1(const _Float16* __restrict__ hp, const _Float16* __restrict__ wh,
           const float* __restrict__ bias, _Float16* __restrict__ out) {
    int w    = blockIdx.x * 8 + (threadIdx.x >> 5);
    int lane = threadIdx.x & 31;
    int mt = w >> 3, nt = w & 7;
    int m  = lane & 15;
    int hb = (lane >> 4) & 1;
    int row = mt * 16 + m;                 // A row (batch)
    int n   = nt * 16 + m;                 // B column (out feature)
    const _Float16* pa = hp + (size_t)row * KFC + hb * 8;
    const _Float16* pb = wh + (size_t)n   * KFC + hb * 8;
    v8f acc = {};
    for (int kb = 0; kb < KFC; kb += 32) {
        Frag16 a, bf;
        a.q[0]  = *(const uint4*)(pa + kb);
        a.q[1]  = *(const uint4*)(pa + kb + 16);
        bf.q[0] = *(const uint4*)(pb + kb);
        bf.q[1] = *(const uint4*)(pb + kb + 16);
        __builtin_prefetch(pa + kb + 128, 0, 0);
        __builtin_prefetch(pb + kb + 128, 0, 0);
        acc = __builtin_amdgcn_wmma_f32_16x16x32_f16(
            false, a.h, false, bf.h, (short)0, acc, false, false);
    }
    float bv = bias[n];
    #pragma unroll
    for (int r = 0; r < 8; ++r) {
        int rr = mt * 16 + r + hb * 8;
        out[(size_t)rr * NFC + n] = (_Float16)fmaxf(acc[r] + bv, 0.f);
    }
}

// FC2 (128->10) + log_softmax; one wave per batch row.
__global__ __launch_bounds__(128)
void k_fc2(const _Float16* __restrict__ h, const float* __restrict__ w2,
           const float* __restrict__ b2, float* __restrict__ out) {
    int b = blockIdx.x * 4 + (threadIdx.x >> 5);
    int l = threadIdx.x & 31;
    float hv[4];
    #pragma unroll
    for (int i = 0; i < 4; ++i) hv[i] = (float)h[(size_t)b * NFC + l + 32 * i];
    float lj[10];
    #pragma unroll
    for (int j = 0; j < 10; ++j) {
        float p = 0.f;
        #pragma unroll
        for (int i = 0; i < 4; ++i) p += hv[i] * w2[j * NFC + l + 32 * i];
        for (int off = 16; off > 0; off >>= 1) p += __shfl_xor(p, off);
        lj[j] = p + b2[j];
    }
    float mx = lj[0];
    #pragma unroll
    for (int j = 1; j < 10; ++j) mx = fmaxf(mx, lj[j]);
    float sum = 0.f;
    #pragma unroll
    for (int j = 0; j < 10; ++j) sum += expf(lj[j] - mx);
    float lse = logf(sum);
    if (l < 10) out[(size_t)b * 10 + l] = lj[l] - mx - lse;
}

// ---------------------------------------------------------------------------
extern "C" void kernel_launch(void* const* d_in, const int* in_sizes, int n_in,
                              void* d_out, int out_size, void* d_ws, size_t ws_size,
                              hipStream_t stream) {
    const float* x     = (const float*)d_in[0];
    const float* gamma = (const float*)d_in[1];
    const float* beta  = (const float*)d_in[2];
    const float* w1    = (const float*)d_in[3];
    const float* b1    = (const float*)d_in[4];
    const float* w2    = (const float*)d_in[5];
    const float* b2    = (const float*)d_in[6];
    const float* fcw1  = (const float*)d_in[7];
    const float* fcb1  = (const float*)d_in[8];
    const float* fcw2  = (const float*)d_in[9];
    const float* fcb2  = (const float*)d_in[10];
    float*       out   = (float*)d_out;

    char* ws = (char*)d_ws;
    float*     stats = (float*)(ws + OFF_STATS);
    _Float16*  h1    = (_Float16*)(ws + OFF_H1);
    _Float16*  hp    = (_Float16*)(ws + OFF_HP);
    _Float16*  fc1o  = (_Float16*)(ws + OFF_FC1);
    _Float16*  fw1h  = (_Float16*)(ws + OFF_FW1);
    uint32_t*  w2p   = (uint32_t*)(ws + OFF_W2P);

    k_init_stats<<<1, 32, 0, stream>>>(stats);
    k_reduce<<<1024, 256, 0, stream>>>(x, stats);
    k_finalize<<<1, 1, 0, stream>>>(stats, gamma, beta);

    k_prep_fcw1<<<(KFC * NFC + 255) / 256, 256, 0, stream>>>(fcw1, fw1h);
    k_prep_w2<<<36, 256, 0, stream>>>(w2, w2p);

    k_bn_conv1<<<(NX + 127) / 128, 128, 0, stream>>>(x, stats, w1, b1, h1);
    k_conv2<<<NB, 128, 0, stream>>>(x, h1, w2p, b2, hp);
    k_fc1<<<256, 256, 0, stream>>>(hp, fw1h, fcb1, fc1o);
    k_fc2<<<NB / 4, 128, 0, stream>>>(fc1o, fcw2, fcb2, out);
}
